// Encoder_2869038154260
// MI455X (gfx1250) — matrix operand order repaired and checked by reading, not compiled, not run on hardware
//
#include <hip/hip_runtime.h>
#include <hip/hip_bf16.h>
#include <math.h>

// ---------------- problem constants ----------------
#define D      64
#define DH     32
#define E_N    64
#define TOPK   4
#define NV     3
#define RES_   5000
#define FEAT_  200
#define NT     2048      // B*T = 128*16
#define D4     256       // 4*D

typedef __bf16 bf16_t;
typedef __attribute__((ext_vector_type(16))) __bf16 v16bf;
typedef __attribute__((ext_vector_type(8)))  __bf16 v8bf;
typedef __attribute__((ext_vector_type(8)))  float  v8f;

// ---- WMMA helper (CDNA5: v_wmma_f32_16x16x32_bf16, wave32) ----
__device__ __forceinline__ v8f wmma_bf16(v16bf a, v16bf b, v8f c) {
  // 8 args: (neg_a, A, neg_b, B, c_mod, C, reuse_a, reuse_b)
  return __builtin_amdgcn_wmma_f32_16x16x32_bf16(false, a, false, b, (short)0, c,
                                                 false, false);
}

// A-fragment (16x32 bf16, M x K): lane holds row m = lane&15.
// lane half kh = lane>>4: elems 0..7 -> K = k0 + 16*kh + i ; elems 8..15 -> K = k0+16*kh+8+i
__device__ __forceinline__ v16bf load_a_frag(const bf16_t* row, int k0, int kh) {
  v8bf lo = *(const v8bf*)(row + k0 + 16 * kh);
  v8bf hi = *(const v8bf*)(row + k0 + 16 * kh + 8);
  v16bf a;
#pragma unroll
  for (int i = 0; i < 8; ++i) { a[i] = lo[i]; a[i + 8] = hi[i]; }
  return a;
}

// ---------------- kernel 0: prep ----------------
// - W1t[e][n][k] = bf16(W1[e][k][n])  (n in [0,256), k in [0,64))   B-frag contiguous
// - W2t[e][n][k] = bf16(W2[e][k][n])  (n in [0,64),  k in [0,256))  B-frag contiguous
// - zero fused accumulator and per-(view,expert) counters
__global__ void k0_prep(const float* __restrict__ W1, const float* __restrict__ W2,
                        bf16_t* __restrict__ W1t, bf16_t* __restrict__ W2t,
                        float* __restrict__ fused, int* __restrict__ counts) {
  int i = blockIdx.x * blockDim.x + threadIdx.x;
  int stride = gridDim.x * blockDim.x;
  const int NW = E_N * D * D4;  // 1048576
  for (int idx = i; idx < NW; idx += stride) {
    int k  = idx & (D - 1);
    int n  = (idx >> 6) & (D4 - 1);
    int e  = idx >> 14;
    W1t[idx] = (bf16_t)W1[(e * D + k) * D4 + n];
    int k2 = idx & (D4 - 1);
    int n2 = (idx >> 8) & (D - 1);
    int e2 = idx >> 14;
    W2t[idx] = (bf16_t)W2[(e2 * D4 + k2) * D + n2];
  }
  for (int idx = i; idx < NT * D; idx += stride) fused[idx] = 0.0f;
  if (i < NV * E_N) counts[i] = 0;
}

// ---------------- kernel 1: per-token embed/PE/proj/router/top-k ----------------
__global__ __launch_bounds__(64) void k1_token(
    const int* __restrict__ src, const float* __restrict__ frac,
    const float* __restrict__ cbfv, const float* __restrict__ W_m2v,
    const float* __restrict__ b_m2v, const float* __restrict__ embs,
    const float* __restrict__ poss, const float* __restrict__ possl,
    const float* __restrict__ projW, const float* __restrict__ projb,
    const float* __restrict__ routerW, const float* __restrict__ ekeys,
    bf16_t* __restrict__ h_bf, float* __restrict__ gate_ws,
    int* __restrict__ pairs, int* __restrict__ counts) {
  int t = blockIdx.x;      // token id
  int j = threadIdx.x;     // 0..63

  __shared__ float xs[D], pel[DH], plg[DH], hh[D], rr[D], lg[E_N];

  // embedding: x_id[j] = (cbfv[s] @ W_m2v + b)[j] * 2^emb_scaler
  int s = src[t];
  float acc = b_m2v[j];
  for (int f = 0; f < FEAT_; ++f) acc += cbfv[s * FEAT_ + f] * W_m2v[f * D + j];
  xs[j] = acc * exp2f(embs[0]);

  // fractional positional encodings (computed analytically; no table)
  float fv = frac[t];
  if (j < DH) {
    float x  = fmaxf(fv, 1.0f / (float)RES_);
    float xf = fminf(x * (float)RES_, (float)RES_);
    int  idx = (int)rintf(xf) - 1;
    idx = idx < 0 ? 0 : (idx > RES_ - 1 ? RES_ - 1 : idx);
    float arg = (float)idx * powf(50.0f, -(float)j / 16.0f);
    pel[j] = (j & 1) ? cosf(arg) : sinf(arg);
  } else {
    int jj   = j - DH;
    float l  = log2f(fv);
    float x  = 0.0025f * l * l;
    x        = fmaxf(x, 1.0f / (float)RES_);
    float xf = fminf(x * (float)RES_, (float)RES_);
    int  idx = (int)rintf(xf) - 1;
    idx = idx < 0 ? 0 : (idx > RES_ - 1 ? RES_ - 1 : idx);
    float arg = (float)idx * powf(50.0f, -(float)jj / 16.0f);
    plg[jj] = (jj & 1) ? cosf(arg) : sinf(arg);
  }
  __syncthreads();

  float pe_sc  = exp2f((1.0f - poss[0]) * (1.0f - poss[0]));
  float ple_sc = exp2f((1.0f - possl[0]) * (1.0f - possl[0]));

  for (int v = 0; v < NV; ++v) {
    // h = view @ projW[v] + projb[v]   (views 1/2 have only 32 nonzero dims)
    const float* pw = projW + (size_t)v * D * D;
    float h = projb[v * D + j];
    if (v == 0)      { for (int c = 0; c < D;  ++c) h += xs[c] * pw[c * D + j]; }
    else if (v == 1) { for (int c = 0; c < DH; ++c) h += pel[c] * pe_sc  * pw[c * D + j]; }
    else             { for (int c = 0; c < DH; ++c) h += plg[c] * ple_sc * pw[(DH + c) * D + j]; }
    hh[j] = h;
    h_bf[((size_t)v * NT + t) * D + j] = (bf16_t)h;
    __syncthreads();

    // r = h @ routerW[v]
    const float* rw = routerW + (size_t)v * D * D;
    float r = 0.0f;
    for (int c = 0; c < D; ++c) r += hh[c] * rw[c * D + j];
    rr[j] = r;
    __syncthreads();

    // logits[e] = -||r - expert_keys[e]||   (thread j handles expert j)
    const float* ek = ekeys + j * D;
    float dd = 0.0f;
    for (int c = 0; c < D; ++c) { float df = rr[c] - ek[c]; dd += df * df; }
    lg[j] = -sqrtf(dd);
    __syncthreads();

    if (j == 0) {
      float tv[TOPK]; int ti[TOPK];
      for (int kk = 0; kk < TOPK; ++kk) {
        float vm = -3.4e38f; int im = 0;
        for (int e = 0; e < E_N; ++e) if (lg[e] > vm) { vm = lg[e]; im = e; }
        tv[kk] = vm; ti[kk] = im; lg[im] = -3.4e38f;
      }
      float mx = tv[0], ssum = 0.0f, ex[TOPK];
      for (int kk = 0; kk < TOPK; ++kk) { ex[kk] = expf(tv[kk] - mx); ssum += ex[kk]; }
      for (int kk = 0; kk < TOPK; ++kk) {
        gate_ws[((size_t)v * NT + t) * TOPK + kk] = ex[kk] / ssum;
        int eo  = v * E_N + ti[kk];
        int pos = atomicAdd(&counts[eo], 1);
        pairs[(size_t)eo * NT + pos] = (t << 2) | kk;
      }
    }
    __syncthreads();
  }
}

// ---------------- kernel 2: sparse grouped expert MLP via WMMA ----------------
// All weight fragments for this (expert, wave) are loaded ONCE into registers:
// 8 W1 B-frags (64 VGPR) + 8 W2 B-frags (64 VGPR). Per tile the only memory
// traffic is the gathered h rows (A-frags), pair/gate metadata, LDS GELU
// staging, and the atomic scatter. Next tile's h rows are prefetched.
__global__ __launch_bounds__(128) void k2_moe(
    const bf16_t* __restrict__ h_bf, const float* __restrict__ gate_ws,
    const int* __restrict__ pairs, const int* __restrict__ counts,
    const bf16_t* __restrict__ W1t, const bf16_t* __restrict__ W2t,
    const float* __restrict__ b1, const float* __restrict__ b2,
    float* __restrict__ fused) {
  int e = blockIdx.x, v = blockIdx.y;
  int cnt = counts[v * E_N + e];
  if (cnt == 0) return;                         // block-uniform
  const int* plist = pairs + (size_t)(v * E_N + e) * NT;

  int lane = threadIdx.x & 31;
  int wave = threadIdx.x >> 5;                  // 4 wave32s
  int m    = lane & 15;
  int kh   = lane >> 4;

  __shared__ __align__(32) bf16_t y1s[16 * D4]; // 8 KB GELU staging (C-frag -> A-frag relayout)
  __shared__ float gts[16];
  __shared__ int   toks[16];

  const bf16_t* w1e = W1t + (size_t)e * D4 * D; // [256][64]
  const bf16_t* w2e = W2t + (size_t)e * D * D4; // [64][256]

  // ---- hoist weight fragments + biases into registers (block-lifetime) ----
  v16bf w1f[8];          // [q][kc] for this wave's 4 N-chunks of GEMM1
  float bias1[4];
#pragma unroll
  for (int q = 0; q < 4; ++q) {
    int n = (wave * 4 + q) * 16 + m;
    bias1[q] = b1[e * D4 + n];
#pragma unroll
    for (int kc = 0; kc < 2; ++kc)
      w1f[q * 2 + kc] = *(const v16bf*)(w1e + (size_t)n * D + kc * 32 + kh * 16);
  }
  int n2 = wave * 16 + m;                       // GEMM2 output column of this lane
  v16bf w2f[8];
#pragma unroll
  for (int kc = 0; kc < 8; ++kc)
    w2f[kc] = *(const v16bf*)(w2e + (size_t)n2 * D4 + kc * 32 + kh * 16);
  float bias2 = b2[e * D + n2];

  int ntiles = (cnt + 15) >> 4;
  for (int tile = 0; tile < ntiles; ++tile) {
    __syncthreads();                            // y1s/toks reuse across tiles
    if (threadIdx.x < 16) {
      int i  = tile * 16 + threadIdx.x;
      int pr = (i < cnt) ? plist[i] : plist[0]; // clamp tail to a valid pair, gate=0
      int tok = pr >> 2;
      toks[threadIdx.x] = tok;
      gts[threadIdx.x]  = (i < cnt)
          ? gate_ws[((size_t)v * NT + tok) * TOPK + (pr & 3)] : 0.0f;
    }
    __syncthreads();

    // prefetch next tile's gathered h rows (global_prefetch_b8; no counter cost)
    {
      int ni = (tile + 1) * 16 + m;
      if (wave == 0 && kh == 0 && ni < cnt) {
        int prn = plist[ni];
        __builtin_prefetch(h_bf + ((size_t)v * NT + (prn >> 2)) * D, 0, 3);
      }
    }

    // ---- GEMM1: [16x64] h-tile x [64x256] W1 (register-resident B) ----
    const bf16_t* hrow = h_bf + ((size_t)v * NT + toks[m]) * D;
    v16bf a0 = load_a_frag(hrow, 0, kh);
    v16bf a1 = load_a_frag(hrow, 32, kh);
#pragma unroll
    for (int q = 0; q < 4; ++q) {
      v8f c = {};
      c = wmma_bf16(a0, w1f[q * 2 + 0], c);
      c = wmma_bf16(a1, w1f[q * 2 + 1], c);
      int n = (wave * 4 + q) * 16 + m;
#pragma unroll
      for (int r = 0; r < 8; ++r) {             // exact GELU, write as GEMM2 A-operand
        float x = c[r] + bias1[q];
        float g = 0.5f * x * (1.0f + erff(x * 0.70710678118654752f));
        y1s[(r + 8 * kh) * D4 + n] = (bf16_t)g;
      }
    }
    __syncthreads();

    // ---- GEMM2: [16x256] gelu-tile x [256x64] W2 (register-resident B) ----
    v8f c2 = {};
#pragma unroll
    for (int kc = 0; kc < 8; ++kc) {
      v16bf a = load_a_frag(&y1s[m * D4], kc * 32, kh);
      c2 = wmma_bf16(a, w2f[kc], c2);
    }
#pragma unroll
    for (int r = 0; r < 8; ++r) {
      int mr = r + 8 * kh;
      atomicAdd(&fused[(size_t)toks[mr] * D + n2], gts[mr] * (c2[r] + bias2));
    }
  }
}

// ---------------- kernel 3: hmask + writeout ----------------
__global__ void k3_final(const float* __restrict__ fused, const float* __restrict__ frac,
                         float* __restrict__ out) {
  int i = blockIdx.x * blockDim.x + threadIdx.x;
  if (i >= NT * D) return;
  int t = i >> 6;
  int b = t >> 4;
  float hm = frac[t] * frac[b * 16];
  out[i] = (hm != 0.0f) ? fused[i] : 0.0f;
}

// ---------------- launcher ----------------
extern "C" void kernel_launch(void* const* d_in, const int* in_sizes, int n_in,
                              void* d_out, int out_size, void* d_ws, size_t ws_size,
                              hipStream_t stream) {
  const int*   src     = (const int*)d_in[0];
  const float* frac    = (const float*)d_in[1];
  const float* cbfv    = (const float*)d_in[2];
  const float* W_m2v   = (const float*)d_in[3];
  const float* b_m2v   = (const float*)d_in[4];
  const float* embs    = (const float*)d_in[5];
  const float* poss    = (const float*)d_in[6];
  const float* possl   = (const float*)d_in[7];
  const float* projW   = (const float*)d_in[8];
  const float* projb   = (const float*)d_in[9];
  const float* routerW = (const float*)d_in[10];
  const float* ekeys   = (const float*)d_in[11];
  const float* W1      = (const float*)d_in[12];
  const float* b1      = (const float*)d_in[13];
  const float* W2      = (const float*)d_in[14];
  const float* b2      = (const float*)d_in[15];

  char* ws = (char*)d_ws;
  float*  fused   = (float*)(ws + 0);            // 2048*64*4      = 524288
  bf16_t* h_bf    = (bf16_t*)(ws + 524288);      // 3*2048*64*2    = 786432
  float*  gate_ws = (float*)(ws + 1310720);      // 3*2048*4*4     = 98304
  int*    counts  = (int*)(ws + 1409024);        // 192*4 (pad 1024)
  int*    pairs   = (int*)(ws + 1410048);        // 3*64*2048*4    = 1572864
  bf16_t* W1t     = (bf16_t*)(ws + 2982912);     // 1M bf16        = 2097152
  bf16_t* W2t     = (bf16_t*)(ws + 5080064);     // 1M bf16        = 2097152
                                                 // total ~7.0 MB

  hipLaunchKernelGGL(k0_prep, dim3(1024), dim3(256), 0, stream,
                     W1, W2, W1t, W2t, fused, counts);
  hipLaunchKernelGGL(k1_token, dim3(NT), dim3(64), 0, stream,
                     src, frac, cbfv, W_m2v, b_m2v, embs, poss, possl,
                     projW, projb, routerW, ekeys, h_bf, gate_ws, pairs, counts);
  hipLaunchKernelGGL(k2_moe, dim3(E_N, NV), dim3(128), 0, stream,
                     h_bf, gate_ws, pairs, counts, W1t, W2t, b1, b2, fused);
  hipLaunchKernelGGL(k3_final, dim3((NT * D + 255) / 256), dim3(256), 0, stream,
                     fused, frac, (float*)d_out);
}
